// MultiHeadSelfAttention_6236292514436
// MI455X (gfx1250) — matrix operand order repaired and checked by reading, not compiled, not run on hardware
//
#include <hip/hip_runtime.h>
#include <math.h>

// ---- CDNA5 wave32 WMMA types -------------------------------------------------
typedef __bf16 bf16;
typedef __attribute__((ext_vector_type(4)))  __bf16 v4bf;
typedef __attribute__((ext_vector_type(8)))  __bf16 v8bf;
typedef __attribute__((ext_vector_type(16))) __bf16 v16bf;
typedef __attribute__((ext_vector_type(8)))  float  v8f;

#define D_MODEL 1024
#define N_ROWS  8192          // B*S = 4*2048
#define SEQ     2048
#define HEADS   16
#define HDIM    64

// sched_group_barrier masks
#define SG_WMMA   0x008
#define SG_VMEM_R 0x020
#define SG_DS_R   0x100
#define SG_DS_W   0x200

static __device__ inline bf16 f2bf(float f) {
  unsigned int u = __builtin_bit_cast(unsigned int, f);
  unsigned int r = (u + 0x7FFFu + ((u >> 16) & 1u)) >> 16;
  unsigned short s = (unsigned short)r;
  return __builtin_bit_cast(bf16, s);
}

static __device__ inline v8f bwmma(v16bf a, v16bf b, v8f c) {
  // D = A(16x32 bf16) x B(32x16 bf16) + C(16x16 f32)
  return __builtin_amdgcn_wmma_f32_16x16x32_bf16(
      false, a, false, b, (short)0, c, false, false);
}

// async global->LDS 16B copy, tracked by ASYNCcnt (CDNA5 ISA)
static __device__ inline void async_cp16(const bf16* gsrc, bf16* lds_dst) {
  unsigned lds = (unsigned)(unsigned long long)(void*)lds_dst; // low 32b = LDS offset
  unsigned long long ga = (unsigned long long)(void*)gsrc;
  asm volatile("global_load_async_to_lds_b128 %0, %1, off"
               :: "v"(lds), "v"(ga) : "memory");
}
static __device__ inline void wait_async0() {
  asm volatile("s_wait_asynccnt 0" ::: "memory");
}

// A fragment (16x32, row-major [ld]): lane = row, elems 0..7 = K kb..kb+7,
// elems 8..15 = K 16+kb..16+kb+7 (kb = 8*(lane>=16)).
static __device__ inline v16bf load_a(const bf16* __restrict__ base, int ld,
                                      int row0, int k0, int lane) {
  int row = row0 + (lane & 15);
  int kb  = (lane >> 4) << 3;
  const bf16* p = base + row * ld + k0;
  union { v16bf v; v8bf h[2]; } u;
  u.h[0] = *(const v8bf*)(p + kb);
  u.h[1] = *(const v8bf*)(p + 16 + kb);
  return u.v;
}

// B fragment (32x16) from a K-contiguous [N][K] layout (row stride = ldk elems)
static __device__ inline v16bf frag_bt(const bf16* bt, int ldk, int n_local,
                                       int k_local, int lane) {
  int col = n_local + (lane & 15);
  int kh  = (lane >> 4) << 3;
  const bf16* p = bt + col * ldk + k_local + kh;
  union { v16bf v; v8bf h[2]; } u;
  u.h[0] = *(const v8bf*)(p);
  u.h[1] = *(const v8bf*)(p + 16);
  return u.v;
}

// ---- prep kernels ------------------------------------------------------------
__global__ __launch_bounds__(256) void cvt_x_bf16(const float* __restrict__ in,
                                                  bf16* __restrict__ out) {
  int i = (blockIdx.x * 256 + threadIdx.x) * 4;
  const float4 f = *(const float4*)(in + i);
  v4bf o; o[0] = f2bf(f.x); o[1] = f2bf(f.y); o[2] = f2bf(f.z); o[3] = f2bf(f.w);
  *(v4bf*)(out + i) = o;
}

__global__ __launch_bounds__(256) void transpose_w(const float* __restrict__ w,
                                                   bf16* __restrict__ wt) {
  int t = blockIdx.x * 256 + threadIdx.x;
  int n = t >> 10;
  int k = t & 1023;
  wt[n * D_MODEL + k] = f2bf(w[k * D_MODEL + n]); // wt[N][K] = W[K][N]
}

// ---- fused projection --------------------------------------------------------
// Block = 8 waves = 128 rows x 64 cols. Weight strip shared via async LDS tiles.
#define BROW 40   // 32 K + 8 pad elems (80B rows, 16B aligned)

__global__ __launch_bounds__(256) void qkv_proj(const bf16* __restrict__ xb,
                                                const bf16* __restrict__ wt,
                                                const float* __restrict__ bias,
                                                bf16* __restrict__ dst,
                                                int vtrans) {
  __shared__ bf16 Bl[2][64 * BROW];                 // 10 KB, double-buffered W^T tile
  const int tid  = threadIdx.x;
  const int lane = tid & 31;
  const int w    = tid >> 5;
  const int mblock = blockIdx.x >> 4;               // 64 row-blocks of 128
  const int strip  = blockIdx.x & 15;               // 16 strips of 64 cols
  const int row0 = mblock * 128 + w * 16;
  const int n0   = strip << 6;

  const int fn = tid >> 2, fc = tid & 3;            // fill decode: 64 rows x 4 chunks
  auto fill = [&](int buf, int k0) {
    async_cp16(wt + (n0 + fn) * D_MODEL + k0 + fc * 8,
               &Bl[buf][fn * BROW + fc * 8]);
  };

  fill(0, 0);

  v8f acc[4];
  #pragma unroll
  for (int t = 0; t < 4; ++t) acc[t] = (v8f){0.f,0.f,0.f,0.f,0.f,0.f,0.f,0.f};

  v16bf a_cur = load_a(xb, D_MODEL, row0, 0, lane);

  for (int k0 = 0; k0 < D_MODEL; k0 += 32) {
    const int buf = (k0 >> 5) & 1;
    wait_async0();
    __syncthreads();                                // tile[buf] ready; prev reads done
    if (k0 + 32 < D_MODEL) fill(buf ^ 1, k0 + 32);  // overlap next fill with math

    v16bf a_next = a_cur;
    if (k0 + 32 < D_MODEL) a_next = load_a(xb, D_MODEL, row0, k0 + 32, lane);

    // issue ALL fragment loads first, then the WMMA chain (partial ds waits)
    const bf16* bl = &Bl[buf][0];
    v16bf bfrag[4];
    #pragma unroll
    for (int nt = 0; nt < 4; ++nt)
      bfrag[nt] = frag_bt(bl, BROW, nt * 16, 0, lane);
    #pragma unroll
    for (int nt = 0; nt < 4; ++nt)
      acc[nt] = bwmma(a_cur, bfrag[nt], acc[nt]);

    // pin schedule: A global reads, then 8 DS reads, then 4 WMMAs
    __builtin_amdgcn_sched_group_barrier(SG_VMEM_R, 2, 0);
    __builtin_amdgcn_sched_group_barrier(SG_DS_R,   8, 0);
    __builtin_amdgcn_sched_group_barrier(SG_WMMA,   4, 0);

    a_cur = a_next;
  }

  const int cl    = lane & 15;
  const int rbase = row0 + ((lane >> 4) << 3);
  #pragma unroll
  for (int nt = 0; nt < 4; ++nt) {
    const int colg = n0 + nt * 16 + cl;
    const float bv = bias[colg];
    if (!vtrans) {
      #pragma unroll
      for (int v = 0; v < 8; ++v)
        dst[(rbase + v) * D_MODEL + colg] = f2bf(acc[nt][v] + bv);
    } else {
      v8bf pk;                                      // V^T: one b128 store per lane
      #pragma unroll
      for (int v = 0; v < 8; ++v) pk[v] = f2bf(acc[nt][v] + bv);
      *(v8bf*)(dst + colg * N_ROWS + rbase) = pk;
    }
  }
}

// ---- flash attention ---------------------------------------------------------
// Block = 4 waves x 16 queries, one head. K/V tiles shared by all waves via
// async LDS double buffers; P transposed through per-wave LDS scratch.
#define KROW 72   // 64 dh + 8 pad (144B rows)
#define VROW 40   // 32 k + 8 pad (80B rows)

__global__ __launch_bounds__(128) void mha_flash(const bf16* __restrict__ Qb,
                                                 const bf16* __restrict__ Kb,
                                                 const bf16* __restrict__ Vt,
                                                 float* __restrict__ out) {
  __shared__ bf16 Kl[2][32 * KROW];                 // [krow][dh]  9.2 KB
  __shared__ bf16 Vl[2][64 * VROW];                 // [dh][k]    10.2 KB
  __shared__ bf16 Pl[4][16 * 32];                   // per-wave P  4.0 KB
  const int tid   = threadIdx.x;
  const int lane  = tid & 31;
  const int w     = tid >> 5;
  const int qrow0 = blockIdx.x * 64 + w * 16;
  const int h     = blockIdx.y;
  const int batch = qrow0 >> 11;
  const int dbase = h << 6;

  const int kfr = tid >> 2, kfc = tid & 3;          // K fill: 32 rows x 4 chunks
  auto fill = [&](int buf, int kr) {
    async_cp16(Kb + (kr + kfr) * D_MODEL + dbase + kfc * 8,
               &Kl[buf][kfr * KROW + kfc * 8]);
    #pragma unroll
    for (int i = 0; i < 2; ++i) {                   // V fill: 64 rows x 4 chunks
      int idx = tid + i * 128;
      int d = idx >> 2, c = idx & 3;
      async_cp16(Vt + (dbase + d) * N_ROWS + kr + c * 8,
                 &Vl[buf][d * VROW + c * 8]);
    }
  };

  const v16bf qa0 = load_a(Qb, D_MODEL, qrow0, dbase,      lane);
  const v16bf qa1 = load_a(Qb, D_MODEL, qrow0, dbase + 32, lane);

  const v8f zero = (v8f){0.f,0.f,0.f,0.f,0.f,0.f,0.f,0.f};
  v8f o0 = zero, o1 = zero, o2 = zero, o3 = zero;
  float m[8], l[8];
  #pragma unroll
  for (int v = 0; v < 8; ++v) { m[v] = -3.0e38f; l[v] = 0.0f; }

  bf16* Pw = Pl[w];
  const int cl = lane & 15;
  const int kb = (lane >> 4) << 3;

  fill(0, (batch << 11));

  for (int kt = 0; kt < SEQ; kt += 32) {
    const int kr  = (batch << 11) + kt;
    const int buf = (kt >> 5) & 1;
    wait_async0();
    __syncthreads();                                // K/V[buf] ready; prev reads done
    if (kt + 32 < SEQ) fill(buf ^ 1, kr + 32);

    // scores: issue all 4 K-fragment loads, then the 4 WMMAs
    const bf16* kl = &Kl[buf][0];
    v16bf kf[4];
    kf[0] = frag_bt(kl, KROW, 0,  0,  lane);
    kf[1] = frag_bt(kl, KROW, 0,  32, lane);
    kf[2] = frag_bt(kl, KROW, 16, 0,  lane);
    kf[3] = frag_bt(kl, KROW, 16, 32, lane);
    v8f s0 = zero, s1 = zero;
    s0 = bwmma(qa0, kf[0], s0);
    s0 = bwmma(qa1, kf[1], s0);
    s1 = bwmma(qa0, kf[2], s1);
    s1 = bwmma(qa1, kf[3], s1);

    // fp32 online softmax; row stats per accumulator row, 16-lane reductions
    #pragma unroll
    for (int v = 0; v < 8; ++v) {
      float x0 = s0[v] * 0.125f;                    // 1/sqrt(64)
      float x1 = s1[v] * 0.125f;
      float t = fmaxf(x0, x1);
      t = fmaxf(t, __shfl_xor(t, 1, 32));
      t = fmaxf(t, __shfl_xor(t, 2, 32));
      t = fmaxf(t, __shfl_xor(t, 4, 32));
      t = fmaxf(t, __shfl_xor(t, 8, 32));
      float mn   = fmaxf(m[v], t);
      float corr = __expf(m[v] - mn);
      float p0   = __expf(x0 - mn);
      float p1   = __expf(x1 - mn);
      float ps   = p0 + p1;
      ps += __shfl_xor(ps, 1, 32);
      ps += __shfl_xor(ps, 2, 32);
      ps += __shfl_xor(ps, 4, 32);
      ps += __shfl_xor(ps, 8, 32);
      l[v] = l[v] * corr + ps;
      m[v] = mn;
      o0[v] *= corr; o1[v] *= corr; o2[v] *= corr; o3[v] *= corr;
      const int rr = v + ((lane >> 4) << 3);
      Pw[rr * 32 + cl]      = f2bf(p0);
      Pw[rr * 32 + 16 + cl] = f2bf(p1);
    }

    // re-read P in A layout (wave-private LDS region, DS ops are in-order);
    // V fragments issued in the same load group so waits are partial
    union { v16bf v; v8bf h[2]; } pu;
    pu.h[0] = *(const v8bf*)&Pw[cl * 32 + kb];
    pu.h[1] = *(const v8bf*)&Pw[cl * 32 + 16 + kb];
    const bf16* vl = &Vl[buf][0];
    v16bf vf[4];
    vf[0] = frag_bt(vl, VROW, 0,  0, lane);
    vf[1] = frag_bt(vl, VROW, 16, 0, lane);
    vf[2] = frag_bt(vl, VROW, 32, 0, lane);
    vf[3] = frag_bt(vl, VROW, 48, 0, lane);
    const v16bf pa = pu.v;

    // O += P . V
    o0 = bwmma(pa, vf[0], o0);
    o1 = bwmma(pa, vf[1], o1);
    o2 = bwmma(pa, vf[2], o2);
    o3 = bwmma(pa, vf[3], o3);

    // pin schedule: Kfrag reads -> score WMMAs -> P writes -> P+V reads -> PV WMMAs
    __builtin_amdgcn_sched_group_barrier(SG_DS_R,  8, 0);
    __builtin_amdgcn_sched_group_barrier(SG_WMMA,  4, 0);
    __builtin_amdgcn_sched_group_barrier(SG_DS_W, 16, 0);
    __builtin_amdgcn_sched_group_barrier(SG_DS_R, 10, 0);
    __builtin_amdgcn_sched_group_barrier(SG_WMMA,  4, 0);
  }

  const int rbase = qrow0 + ((lane >> 4) << 3);
  #pragma unroll
  for (int v = 0; v < 8; ++v) {
    const float inv = 1.0f / l[v];
    float* orow = out + (rbase + v) * D_MODEL + dbase + cl;
    orow[0]  = o0[v] * inv;
    orow[16] = o1[v] * inv;
    orow[32] = o2[v] * inv;
    orow[48] = o3[v] * inv;
  }
}

// ---- host glue ---------------------------------------------------------------
extern "C" void kernel_launch(void* const* d_in, const int* in_sizes, int n_in,
                              void* d_out, int out_size, void* d_ws, size_t ws_size,
                              hipStream_t stream) {
  const float* x  = (const float*)d_in[0];
  const float* Wq = (const float*)d_in[1];
  const float* bq = (const float*)d_in[2];
  const float* Wk = (const float*)d_in[3];
  const float* bk = (const float*)d_in[4];
  const float* Wv = (const float*)d_in[5];
  const float* bv = (const float*)d_in[6];
  float* out = (float*)d_out;

  char* ws = (char*)d_ws;
  const size_t XB = (size_t)N_ROWS * D_MODEL * sizeof(bf16);   // 16.78 MB
  const size_t WT = (size_t)D_MODEL * D_MODEL * sizeof(bf16);  //  2.10 MB
  bf16* xb  = (bf16*)(ws);
  bf16* wqt = (bf16*)(ws + XB);
  bf16* wkt = (bf16*)(ws + XB + WT);
  bf16* wvt = (bf16*)(ws + XB + 2 * WT);
  bf16* Qb  = (bf16*)(ws + XB + 3 * WT);
  bf16* Kb  = (bf16*)(ws + XB + 3 * WT + XB);
  bf16* Vt  = (bf16*)(ws + XB + 3 * WT + 2 * XB);              // end ~73.4 MB

  cvt_x_bf16<<<(N_ROWS * D_MODEL) / (256 * 4), 256, 0, stream>>>(x, xb);

  transpose_w<<<(D_MODEL * D_MODEL) / 256, 256, 0, stream>>>(Wq, wqt);
  transpose_w<<<(D_MODEL * D_MODEL) / 256, 256, 0, stream>>>(Wk, wkt);
  transpose_w<<<(D_MODEL * D_MODEL) / 256, 256, 0, stream>>>(Wv, wvt);

  qkv_proj<<<1024, 256, 0, stream>>>(xb, wqt, bq, Qb, 0);
  qkv_proj<<<1024, 256, 0, stream>>>(xb, wkt, bk, Kb, 0);
  qkv_proj<<<1024, 256, 0, stream>>>(xb, wvt, bv, Vt, 1);      // V stored transposed

  mha_flash<<<dim3(N_ROWS / 64, HEADS), 128, 0, stream>>>(Qb, Kb, Vt, out);
}
